// DecoderAttentionRNN_42021960024658
// MI455X (gfx1250) — compile-verified, matrix operand order checked by reading, not run on hardware
//
#include <hip/hip_runtime.h>

typedef __attribute__((ext_vector_type(16))) __bf16 v16bf;
typedef __attribute__((ext_vector_type(8)))  __bf16 v8bf;
typedef __attribute__((ext_vector_type(8)))  float  v8f;

#define L_  2
#define H_  512
#define A_  200
#define AP_ 208   // A padded to a multiple of 16 for WMMA tiles
#define E_  512
#define V_  10000
#define B_  64
#define T_  512
#define S_  128

// ---------------- conversion / packing kernels ----------------

__global__ void cvt_bf16_kernel(__bf16* __restrict__ dst, const float* __restrict__ src, long n) {
  long i = (long)blockIdx.x * blockDim.x + threadIdx.x;
  long stride = (long)gridDim.x * blockDim.x;
  for (; i < n; i += stride) dst[i] = (__bf16)src[i];
}

// (L,A,H) f32 -> (L,AP,H) bf16, zero-padded rows A..AP-1
__global__ void pad_w_kernel(__bf16* __restrict__ dst, const float* __restrict__ src) {
  int n = L_ * AP_ * H_;
  for (int i = blockIdx.x * blockDim.x + threadIdx.x; i < n; i += gridDim.x * blockDim.x) {
    int l = i / (AP_ * H_);
    int r = i % (AP_ * H_);
    int a = r / H_, h = r % H_;
    dst[i] = (a < A_) ? (__bf16)src[((size_t)l * A_ + a) * H_ + h] : (__bf16)0.0f;
  }
}

__global__ void pad_b_kernel(float* __restrict__ dst, const float* __restrict__ src) {
  int i = blockIdx.x * blockDim.x + threadIdx.x;
  if (i < L_ * AP_) {
    int l = i / AP_, a = i % AP_;
    dst[i] = (a < A_) ? src[l * A_ + a] : 0.0f;
  }
}

__global__ void init_h_kernel(float* __restrict__ hf, __bf16* __restrict__ hb,
                              const float* __restrict__ h0) {
  int i = blockIdx.x * blockDim.x + threadIdx.x;
  if (i < L_ * B_ * H_) {
    float v = h0[i];
    hf[i] = v;
    hb[i] = (__bf16)v;
  }
}

// ---------------- WMMA GEMM: C[M,N] = A[M,K] @ W[N,K]^T + bias ----------------
// A, W bf16 row-major, leading dimension == K (compile-time) so all k-loop
// loads are base + immediate offset. Each wave computes MREP consecutive
// 16-row tiles for ONE 16-column tile (B fragment loaded once per MREP WMMAs),
// with explicit double-buffering: the next k-step's fragments are loaded
// before the current step's WMMAs so a full iteration of load latency stays
// in flight. __launch_bounds__(256,1) lifts the occupancy-driven VGPR cap so
// the allocator keeps both fragment sets live instead of single-buffering.
// Fragment layout per CDNA5 ISA (16-bit A/B): lane%16 selects row, lane/16
// selects K-half {0..7,16..23} vs {8..15,24..31}.

template <typename OutT, int MREP, int K>
__global__ __launch_bounds__(256, 1) void gemm_bf16_kernel(
    const __bf16* __restrict__ A, const __bf16* __restrict__ W,
    const float* __restrict__ bias, OutT* __restrict__ C,
    int mTiles, int nTiles, long ldc) {
  int wid = blockIdx.x * (blockDim.x >> 5) + (threadIdx.x >> 5);
  int mGroups = mTiles / MREP;
  if (wid >= mGroups * nTiles) return;          // wave-uniform: EXEC stays all-ones
  int mg = wid / nTiles, nt = wid % nTiles;
  int lane = threadIdx.x & 31;
  int hsel = (lane >> 4) * 8;

  const __bf16* pa = A + (size_t)(mg * MREP * 16 + (lane & 15)) * K + hsel;
  const __bf16* pw = W + (size_t)(nt * 16 + (lane & 15)) * K + hsel;

  auto load_frags = [&](int k, v16bf& bv, v16bf* av) {
    v8bf b0 = *(const v8bf*)(pw + k);
    v8bf b1 = *(const v8bf*)(pw + k + 16);
#pragma unroll
    for (int r = 0; r < MREP; ++r) {
      const __bf16* par = pa + r * (16 * K) + k;
      v8bf a0 = *(const v8bf*)par;
      v8bf a1 = *(const v8bf*)(par + 16);
      av[r] = __builtin_shufflevector(a0, a1, 0,1,2,3,4,5,6,7,8,9,10,11,12,13,14,15);
    }
    bv = __builtin_shufflevector(b0, b1, 0,1,2,3,4,5,6,7,8,9,10,11,12,13,14,15);
  };

  v8f zero = {0.f, 0.f, 0.f, 0.f, 0.f, 0.f, 0.f, 0.f};
  v8f acc[MREP];
#pragma unroll
  for (int r = 0; r < MREP; ++r) acc[r] = zero;

  v16bf bvc, avc[MREP];
  load_frags(0, bvc, avc);

#pragma unroll 2
  for (int k = 32; k < K; k += 32) {
    v16bf bvn, avn[MREP];
    load_frags(k, bvn, avn);                    // next step's loads go in flight
#pragma unroll
    for (int r = 0; r < MREP; ++r)
      acc[r] = __builtin_amdgcn_wmma_f32_16x16x32_bf16(false, avc[r], false, bvc, (short)0,
                                                       acc[r], false, false);
    bvc = bvn;
#pragma unroll
    for (int r = 0; r < MREP; ++r) avc[r] = avn[r];
  }
#pragma unroll
  for (int r = 0; r < MREP; ++r)                // epilogue: last k-step
    acc[r] = __builtin_amdgcn_wmma_f32_16x16x32_bf16(false, avc[r], false, bvc, (short)0,
                                                     acc[r], false, false);

  int n = nt * 16 + (lane & 15);
  float bb = bias[n];
#pragma unroll
  for (int r = 0; r < MREP; ++r) {
    int m0 = (mg * MREP + r) * 16 + (lane >> 4) * 8;  // C/D layout: VGPR i -> M = m0 + i
#pragma unroll
    for (int i = 0; i < 8; ++i)
      C[(size_t)(m0 + i) * ldc + n] = (OutT)(acc[r][i] + bb);
  }
}

// ---------------- attention score: s[l,t,b] = Vb[l] + sum_a Vw[l,a]*tanh(q+kp) ----------------
// One wave per (l,t,b); each lane owns 8 contiguous 'a' values so every k_proj
// access is a single 16-byte (b128) load. 25 of 32 lanes active (A=200).

__global__ void attn_scores_kernel(const float* __restrict__ q, const __bf16* __restrict__ kp,
                                   const float* __restrict__ Vw, const float* __restrict__ Vb,
                                   float* __restrict__ sc) {
  int wid = blockIdx.x * (blockDim.x >> 5) + (threadIdx.x >> 5);
  if (wid >= L_ * T_ * B_) return;
  int lane = threadIdx.x & 31;
  int b = wid % B_;
  int lt = wid / B_;
  int l = lt / T_;
  float acc = 0.f;
  if (lane < A_ / 8) {
    const __bf16* kr = kp + (size_t)wid * AP_ + lane * 8;
    const float*  qr = q + (size_t)(l * B_ + b) * AP_ + lane * 8;
    const float*  vr = Vw + l * A_ + lane * 8;
    v8bf kv = *(const v8bf*)kr;
#pragma unroll
    for (int j = 0; j < 8; ++j)
      acc += vr[j] * tanhf(qr[j] + (float)kv[j]);
  }
  for (int off = 16; off; off >>= 1) acc += __shfl_xor(acc, off, 32);
  if (lane == 0) sc[(size_t)lt * B_ + b] = acc + Vb[l];
}

// ---------------- softmax over joint (L*T) per batch column ----------------

__global__ void softmax_kernel(const float* __restrict__ sc, float* __restrict__ w) {
  __shared__ float red[256];
  int b = blockIdx.x, tid = threadIdx.x;
  const int N = L_ * T_;
  float mx = -3.4e38f;
  for (int i = tid; i < N; i += 256) mx = fmaxf(mx, sc[(size_t)i * B_ + b]);
  red[tid] = mx; __syncthreads();
  for (int s = 128; s; s >>= 1) { if (tid < s) red[tid] = fmaxf(red[tid], red[tid + s]); __syncthreads(); }
  mx = red[0]; __syncthreads();
  float sum = 0.f;
  for (int i = tid; i < N; i += 256) {
    float e = expf(sc[(size_t)i * B_ + b] - mx);
    w[(size_t)i * B_ + b] = e;
    sum += e;
  }
  red[tid] = sum; __syncthreads();
  for (int s = 128; s; s >>= 1) { if (tid < s) red[tid] += red[tid + s]; __syncthreads(); }
  float inv = 1.f / red[0];
  __syncthreads();
  for (int i = tid; i < N; i += 256) w[(size_t)i * B_ + b] *= inv;
}

// ---------------- context[b,h] = sum_lt w[lt,b] * EO[lt,b,h] ----------------

__global__ void context_kernel(const float* __restrict__ w, const float* __restrict__ EO,
                               float* __restrict__ ctx) {
  __shared__ float wsm[L_ * T_];
  int b = blockIdx.x, tid = threadIdx.x;
  for (int i = tid; i < L_ * T_; i += 256) wsm[i] = w[(size_t)i * B_ + b];
  __syncthreads();
  float a0 = 0.f, a1 = 0.f;
  for (int lt = 0; lt < L_ * T_; ++lt) {
    const float* row = EO + ((size_t)lt * B_ + b) * H_;
    float wv = wsm[lt];
    a0 += wv * row[tid];
    a1 += wv * row[tid + 256];
  }
  ctx[(size_t)b * H_ + tid] = a0;
  ctx[(size_t)b * H_ + tid + 256] = a1;
}

// ---------------- x = [relu(emb[tok]) , context] in bf16 ----------------

__global__ void embed_concat_kernel(const int* __restrict__ targets, const float* __restrict__ emb,
                                    const float* __restrict__ ctx, __bf16* __restrict__ xbf, int s) {
  int i = blockIdx.x * blockDim.x + threadIdx.x;
  if (i >= B_ * (E_ + H_)) return;
  int b = i >> 10, j = i & 1023;
  int tok = (s == 0) ? 0 : targets[b * S_ + (s - 1)];
  float v = (j < E_) ? fmaxf(emb[(size_t)tok * E_ + j], 0.f)
                     : ctx[(size_t)b * H_ + (j - E_)];
  xbf[i] = (__bf16)v;
}

// ---------------- GRU gate nonlinearity; updates h in f32 + bf16 copies ----------------

__global__ void gru_gate_kernel(const float* __restrict__ Gi, const float* __restrict__ Gh,
                                float* __restrict__ hf, __bf16* __restrict__ hb) {
  int i = blockIdx.x * blockDim.x + threadIdx.x;
  if (i >= B_ * H_) return;
  int b = i >> 9, j = i & 511;
  size_t g = (size_t)b * 3 * H_;
  float ir = Gi[g + j], iz = Gi[g + H_ + j], inn = Gi[g + 2 * H_ + j];
  float hr = Gh[g + j], hz = Gh[g + H_ + j], hn  = Gh[g + 2 * H_ + j];
  float h = hf[i];
  float r = 1.f / (1.f + expf(-(ir + hr)));
  float z = 1.f / (1.f + expf(-(iz + hz)));
  float n = tanhf(inn + r * hn);
  float o = (1.f - z) * n + z * h;
  hf[i] = o;
  hb[i] = (__bf16)o;
}

// ---------------- host orchestration ----------------

static inline int gemm_blocks(int mTiles, int mrep, int nTiles) {
  int waves = (mTiles / mrep) * nTiles;
  return (waves + 7) / 8;
}

extern "C" void kernel_launch(void* const* d_in, const int* in_sizes, int n_in,
                              void* d_out, int out_size, void* d_ws, size_t ws_size,
                              hipStream_t stream) {
  const float* EO   = (const float*)d_in[0];
  const float* h0   = (const float*)d_in[1];
  const int*   tgt  = (const int*)d_in[2];
  const float* Qw   = (const float*)d_in[3];
  const float* Qb   = (const float*)d_in[4];
  const float* Kw   = (const float*)d_in[5];
  const float* Kb   = (const float*)d_in[6];
  const float* Vw   = (const float*)d_in[7];
  const float* Vb   = (const float*)d_in[8];
  const float* emb  = (const float*)d_in[9];
  const float* Wih0 = (const float*)d_in[10];
  const float* Whh0 = (const float*)d_in[11];
  const float* bih0 = (const float*)d_in[12];
  const float* bhh0 = (const float*)d_in[13];
  const float* Wih1 = (const float*)d_in[14];
  const float* Whh1 = (const float*)d_in[15];
  const float* bih1 = (const float*)d_in[16];
  const float* bhh1 = (const float*)d_in[17];
  const float* Pw   = (const float*)d_in[18];
  const float* Pb   = (const float*)d_in[19];
  float* out = (float*)d_out;
  (void)in_sizes; (void)n_in; (void)out_size; (void)ws_size;

  char* p = (char*)d_ws;
  auto carve = [&](size_t bytes) -> char* {
    char* r = p;
    p += (bytes + 255) & ~(size_t)255;
    return r;
  };
  __bf16* EObf   = (__bf16*)carve((size_t)L_ * T_ * B_ * H_ * 2);
  __bf16* kproj  = (__bf16*)carve((size_t)L_ * T_ * B_ * AP_ * 2);   // bf16 -> L2-resident
  __bf16* Pwbf   = (__bf16*)carve((size_t)V_ * H_ * 2);
  __bf16* Wih0bf = (__bf16*)carve((size_t)3 * H_ * (E_ + H_) * 2);
  __bf16* Whh0bf = (__bf16*)carve((size_t)3 * H_ * H_ * 2);
  __bf16* Wih1bf = (__bf16*)carve((size_t)3 * H_ * H_ * 2);
  __bf16* Whh1bf = (__bf16*)carve((size_t)3 * H_ * H_ * 2);
  __bf16* Qwp    = (__bf16*)carve((size_t)L_ * AP_ * H_ * 2);
  __bf16* Kwp    = (__bf16*)carve((size_t)L_ * AP_ * H_ * 2);
  float*  Qbp    = (float*) carve((size_t)L_ * AP_ * 4);
  float*  Kbp    = (float*) carve((size_t)L_ * AP_ * 4);
  float*  hf     = (float*) carve((size_t)L_ * B_ * H_ * 4);
  __bf16* hb     = (__bf16*)carve((size_t)L_ * B_ * H_ * 2);
  float*  qbuf   = (float*) carve((size_t)L_ * B_ * AP_ * 4);
  float*  sc     = (float*) carve((size_t)L_ * T_ * B_ * 4);
  float*  wts    = (float*) carve((size_t)L_ * T_ * B_ * 4);
  float*  ctx    = (float*) carve((size_t)B_ * H_ * 4);
  __bf16* xbf    = (__bf16*)carve((size_t)B_ * (E_ + H_) * 2);
  float*  Gi     = (float*) carve((size_t)B_ * 3 * H_ * 4);
  float*  Gh     = (float*) carve((size_t)B_ * 3 * H_ * 4);

  // one-time conversions (every launch; deterministic)
  cvt_bf16_kernel<<<2048, 256, 0, stream>>>(EObf, EO, (long)L_ * T_ * B_ * H_);
  cvt_bf16_kernel<<<1024, 256, 0, stream>>>(Pwbf, Pw, (long)V_ * H_);
  cvt_bf16_kernel<<<512, 256, 0, stream>>>(Wih0bf, Wih0, (long)3 * H_ * (E_ + H_));
  cvt_bf16_kernel<<<512, 256, 0, stream>>>(Whh0bf, Whh0, (long)3 * H_ * H_);
  cvt_bf16_kernel<<<512, 256, 0, stream>>>(Wih1bf, Wih1, (long)3 * H_ * H_);
  cvt_bf16_kernel<<<512, 256, 0, stream>>>(Whh1bf, Whh1, (long)3 * H_ * H_);
  pad_w_kernel<<<512, 256, 0, stream>>>(Qwp, Qw);
  pad_w_kernel<<<512, 256, 0, stream>>>(Kwp, Kw);
  pad_b_kernel<<<2, 256, 0, stream>>>(Qbp, Qb);
  pad_b_kernel<<<2, 256, 0, stream>>>(Kbp, Kb);
  init_h_kernel<<<(L_ * B_ * H_ + 255) / 256, 256, 0, stream>>>(hf, hb, h0);

  // one-time: k_proj[l] = EO[l] @ Kw[l]^T + Kb  (M=T*B, N=AP, K=H), bf16 output
  for (int l = 0; l < L_; ++l) {
    int mT = (T_ * B_) / 16, nT = AP_ / 16;
    gemm_bf16_kernel<__bf16, 4, H_><<<gemm_blocks(mT, 4, nT), 256, 0, stream>>>(
        EObf + (size_t)l * T_ * B_ * H_, Kwp + (size_t)l * AP_ * H_, Kbp + l * AP_,
        kproj + (size_t)l * T_ * B_ * AP_, mT, nT, (long)AP_);
  }

  for (int s = 0; s < S_; ++s) {
    // q[l] = h[l] @ Qw[l]^T + Qb
    for (int l = 0; l < L_; ++l) {
      gemm_bf16_kernel<float, 2, H_><<<gemm_blocks(B_ / 16, 2, AP_ / 16), 256, 0, stream>>>(
          hb + (size_t)l * B_ * H_, Qwp + (size_t)l * AP_ * H_, Qbp + l * AP_,
          qbuf + (size_t)l * B_ * AP_, B_ / 16, AP_ / 16, (long)AP_);
    }
    attn_scores_kernel<<<(L_ * T_ * B_) / 8, 256, 0, stream>>>(qbuf, kproj, Vw, Vb, sc);
    softmax_kernel<<<B_, 256, 0, stream>>>(sc, wts);
    context_kernel<<<B_, 256, 0, stream>>>(wts, EO, ctx);
    embed_concat_kernel<<<(B_ * (E_ + H_)) / 256, 256, 0, stream>>>(tgt, emb, ctx, xbf, s);

    // GRU layer 0
    gemm_bf16_kernel<float, 2, E_ + H_><<<gemm_blocks(B_ / 16, 2, 3 * H_ / 16), 256, 0, stream>>>(
        xbf, Wih0bf, bih0, Gi, B_ / 16, 3 * H_ / 16, (long)(3 * H_));
    gemm_bf16_kernel<float, 2, H_><<<gemm_blocks(B_ / 16, 2, 3 * H_ / 16), 256, 0, stream>>>(
        hb, Whh0bf, bhh0, Gh, B_ / 16, 3 * H_ / 16, (long)(3 * H_));
    gru_gate_kernel<<<(B_ * H_) / 256, 256, 0, stream>>>(Gi, Gh, hf, hb);

    // GRU layer 1 (input = new h0; hidden = old h1)
    gemm_bf16_kernel<float, 2, H_><<<gemm_blocks(B_ / 16, 2, 3 * H_ / 16), 256, 0, stream>>>(
        hb, Wih1bf, bih1, Gi, B_ / 16, 3 * H_ / 16, (long)(3 * H_));
    gemm_bf16_kernel<float, 2, H_><<<gemm_blocks(B_ / 16, 2, 3 * H_ / 16), 256, 0, stream>>>(
        hb + (size_t)B_ * H_, Whh1bf, bhh1, Gh, B_ / 16, 3 * H_ / 16, (long)(3 * H_));
    gru_gate_kernel<<<(B_ * H_) / 256, 256, 0, stream>>>(Gi, Gh, hf + (size_t)B_ * H_,
                                                         hb + (size_t)B_ * H_);

    // logits[:, s, :] = h1 @ Pw^T + Pb   (C row stride = S*V); B fragment reused 4x
    gemm_bf16_kernel<float, 4, H_><<<gemm_blocks(B_ / 16, 4, V_ / 16), 256, 0, stream>>>(
        hb + (size_t)B_ * H_, Pwbf, Pb, out + (size_t)s * V_, B_ / 16, V_ / 16,
        (long)S_ * V_);
  }
}
